// EdgeBlockCUGOConcat_14027363189336
// MI455X (gfx1250) — matrix operand order repaired and checked by reading, not compiled, not run on hardware
//
#include <hip/hip_runtime.h>
#include <hip/hip_bf16.h>

typedef __bf16 v16bf __attribute__((ext_vector_type(16)));
typedef __bf16 v8bf  __attribute__((ext_vector_type(8)));
typedef float  v8f   __attribute__((ext_vector_type(8)));
typedef float  v4f   __attribute__((ext_vector_type(4)));

#define FD      128      // node/edge feature dim == hidden dim
#define CIN     384      // 3*FD
#define MW      32       // edge rows per wave (2 x 16 WMMA tiles)
#define TILE_M  256      // edges per block (8 waves x 32 rows)
#define WAVES   8
#define RS1     392      // padded LDS row stride (bf16) for W1^T [n][k] (bank-friendly, 16B-chunk aligned)
#define RS2     136      // padded LDS row stride for W2^T [n][k]
#define HRS     136      // padded h-stage row stride (bf16)
#define SRS     132      // padded out-stage row stride (f32)
#define STAGE_BYTES 16896 // per-wave stage: max(32*HRS*2=8704, 32*SRS*4=16896)

#define SM_W1_BYTES (FD * RS1 * 2)        // 100352
#define SM_W2_BYTES (FD * RS2 * 2)        // 34816
#define SM_VEC_BYTES (4 * FD * 4)         // b1,b2,g,b -> 2048
#define SM_TOTAL (SM_W1_BYTES + SM_W2_BYTES + SM_VEC_BYTES + WAVES * STAGE_BYTES) // 272384

static __device__ __forceinline__ v16bf join16(v8bf lo, v8bf hi) {
    v16bf r;
#pragma unroll
    for (int j = 0; j < 8; ++j) { r[j] = lo[j]; r[8 + j] = hi[j]; }
    return r;
}

// A fragment (16-bit A 16x32 ISA layout): base already offset by (s&3)*32 + 8*kh
static __device__ __forceinline__ v16bf cvtA(const float* base) {
    v4f f0 = *(const v4f*)(base);        // K = 8kh + 0..3
    v4f f1 = *(const v4f*)(base + 4);    // K = 8kh + 4..7
    v4f f2 = *(const v4f*)(base + 16);   // K = 16 + 8kh + 0..3
    v4f f3 = *(const v4f*)(base + 20);   // K = 16 + 8kh + 4..7
    v16bf v;
#pragma unroll
    for (int j = 0; j < 4; ++j) {
        v[j]      = (__bf16)f0[j];
        v[4 + j]  = (__bf16)f1[j];
        v[8 + j]  = (__bf16)f2[j];
        v[12 + j] = (__bf16)f3[j];
    }
    return v;
}

// B fragment: lane carries N=16t+mrow, K = 32s + 8kh + 0..7 and 32s + 16 + 8kh + 0..7 (matches A table layout)
static __device__ __forceinline__ v16bf loadB(const __bf16* Wt, int rs, int s, int t,
                                              int mrow, int kh) {
    const __bf16* p = Wt + (16 * t + mrow) * rs + 32 * s + 8 * kh;
    return join16(*(const v8bf*)p, *(const v8bf*)(p + 16));
}

__launch_bounds__(256, 1)
__global__ void edge_mlp_wmma_kernel(const float* __restrict__ efeat,
                                     const float* __restrict__ nfeat,
                                     const int*   __restrict__ src_idx,
                                     const int*   __restrict__ dst_idx,
                                     const float* __restrict__ W1,
                                     const float* __restrict__ b1,
                                     const float* __restrict__ W2,
                                     const float* __restrict__ b2,
                                     const float* __restrict__ ln_g,
                                     const float* __restrict__ ln_b,
                                     float* __restrict__ out,
                                     long long nE, long long nTiles)
{
    extern __shared__ unsigned char smem[];
    __bf16* sW1t = (__bf16*)smem;                               // [FD][RS1] : W1^T (n-major)
    __bf16* sW2t = (__bf16*)(smem + SM_W1_BYTES);               // [FD][RS2] : W2^T (n-major)
    float*  sB1  = (float*)(smem + SM_W1_BYTES + SM_W2_BYTES);
    float*  sB2  = sB1 + FD;
    float*  sG   = sB2 + FD;
    float*  sBt  = sG + FD;
    unsigned char* stageBase = smem + SM_W1_BYTES + SM_W2_BYTES + SM_VEC_BYTES;

    const int tid = threadIdx.x;

    // ---- stage weights transposed (f32 -> bf16), coalesced global reads ----
    for (int i = tid; i < CIN * FD; i += 256) {
        int k = i >> 7, n = i & 127;               // i = k*128 + n
        sW1t[n * RS1 + k] = (__bf16)W1[i];
    }
    for (int i = tid; i < FD * FD; i += 256) {
        int k = i >> 7, n = i & 127;
        sW2t[n * RS2 + k] = (__bf16)W2[i];
    }
    if (tid < FD) { sB1[tid] = b1[tid]; sB2[tid] = b2[tid]; sG[tid] = ln_g[tid]; sBt[tid] = ln_b[tid]; }
    __syncthreads();

    const int wave = tid >> 5;       // wave32: 8 waves per block
    const int lane = tid & 31;
    const int mrow = lane & 15;      // A row index / B,C column (N) index
    const int kh   = lane >> 4;      // which K-half this lane carries
    float*  myStage = (float*)(stageBase + wave * STAGE_BYTES);
    __bf16* myH     = (__bf16*)myStage;

    for (long long tile = blockIdx.x; tile < nTiles; tile += gridDim.x) {
        const long long eBase = tile * TILE_M + (long long)wave * MW;
        long long eL = eBase + mrow;            // rows 0..15
        long long eH = eBase + 16 + mrow;       // rows 16..31
        if (eL >= nE) eL = nE - 1;              // clamp: EXEC stays full for WMMA
        if (eH >= nE) eH = nE - 1;

        const float* r0L = efeat + eL * FD;
        const float* r1L = nfeat + (long long)src_idx[eL] * FD;
        const float* r2L = nfeat + (long long)dst_idx[eL] * FD;
        const float* r0H = efeat + eH * FD;
        const float* r1H = nfeat + (long long)src_idx[eH] * FD;
        const float* r2H = nfeat + (long long)dst_idx[eH] * FD;

        {   // prefetch next tile's efeat rows into caches
            long long nt = tile + gridDim.x;
            if (nt < nTiles) {
                long long pe = nt * TILE_M + (long long)wave * MW + mrow;
                if (pe < nE) __builtin_prefetch(efeat + pe * FD, 0, 0);
                if (pe + 16 < nE) __builtin_prefetch(efeat + (pe + 16) * FD, 0, 0);
            }
        }

        // ---- A fragments, GEMM1: 12 k-steps of 32, two 16-row tiles ----
        v16bf a1L[12], a1H[12];
#pragma unroll
        for (int s = 0; s < 12; ++s) {
            const int off = (s & 3) * 32 + 8 * kh;
            a1L[s] = cvtA((s < 4 ? r0L : (s < 8 ? r1L : r2L)) + off);
            a1H[s] = cvtA((s < 4 ? r0H : (s < 8 ? r1H : r2H)) + off);
        }

        // ---- GEMM1: cat[32,384] @ W1[384,128]; B double-buffered, 2 WMMAs per B ----
        v8f accL[8], accH[8];
#pragma unroll
        for (int t = 0; t < 8; ++t)
#pragma unroll
            for (int j = 0; j < 8; ++j) { accL[t][j] = 0.0f; accH[t][j] = 0.0f; }

        v16bf bcur = loadB(sW1t, RS1, 0, 0, mrow, kh);
#pragma unroll
        for (int i = 0; i < 96; ++i) {          // i = s*8 + t
            const int s = i >> 3, t = i & 7;
            v16bf bnxt = bcur;
            if (i < 95) bnxt = loadB(sW1t, RS1, (i + 1) >> 3, (i + 1) & 7, mrow, kh);
            accL[t] = __builtin_amdgcn_wmma_f32_16x16x32_bf16(
                false, a1L[s], false, bcur, (short)0, accL[t], false, false);
            accH[t] = __builtin_amdgcn_wmma_f32_16x16x32_bf16(
                false, a1H[s], false, bcur, (short)0, accH[t], false, false);
            bcur = bnxt;
        }

        // ---- +b1, SiLU, stage h as bf16 (row-major, padded) ----
#pragma unroll
        for (int t = 0; t < 8; ++t) {
            const float bias = sB1[mrow + 16 * t];
#pragma unroll
            for (int r = 0; r < 8; ++r) {
                float xL = accL[t][r] + bias;
                float xH = accH[t][r] + bias;
                float sL = xL / (1.0f + __expf(-xL));
                float sH = xH / (1.0f + __expf(-xH));
                myH[(r + 8 * kh) * HRS + (mrow + 16 * t)]        = (__bf16)sL;  // rows 0..15
                myH[(16 + r + 8 * kh) * HRS + (mrow + 16 * t)]   = (__bf16)sH;  // rows 16..31
            }
        }
        asm volatile("s_wait_dscnt 0x0" ::: "memory");   // intra-wave LDS RAW fence

        // ---- A fragments, GEMM2 (re-stripe C-layout -> A-layout via LDS) ----
        v16bf a2L[4], a2H[4];
#pragma unroll
        for (int s = 0; s < 4; ++s) {
            const __bf16* hL = myH + mrow * HRS + s * 32 + 8 * kh;
            const __bf16* hH = myH + (16 + mrow) * HRS + s * 32 + 8 * kh;
            a2L[s] = join16(*(const v8bf*)hL, *(const v8bf*)(hL + 16));
            a2H[s] = join16(*(const v8bf*)hH, *(const v8bf*)(hH + 16));
        }
        asm volatile("s_wait_dscnt 0x0" ::: "memory");   // h reads done before stage reuse as f32

        // ---- GEMM2: h[32,128] @ W2[128,128] ----
        v8f acc2L[8], acc2H[8];
#pragma unroll
        for (int t = 0; t < 8; ++t)
#pragma unroll
            for (int j = 0; j < 8; ++j) { acc2L[t][j] = 0.0f; acc2H[t][j] = 0.0f; }

        bcur = loadB(sW2t, RS2, 0, 0, mrow, kh);
#pragma unroll
        for (int i = 0; i < 32; ++i) {          // i = s*8 + t
            const int s = i >> 3, t = i & 7;
            v16bf bnxt = bcur;
            if (i < 31) bnxt = loadB(sW2t, RS2, (i + 1) >> 3, (i + 1) & 7, mrow, kh);
            acc2L[t] = __builtin_amdgcn_wmma_f32_16x16x32_bf16(
                false, a2L[s], false, bcur, (short)0, acc2L[t], false, false);
            acc2H[t] = __builtin_amdgcn_wmma_f32_16x16x32_bf16(
                false, a2H[s], false, bcur, (short)0, acc2H[t], false, false);
            bcur = bnxt;
        }

        // ---- +b2, stage f32 for row-wise LayerNorm ----
#pragma unroll
        for (int t = 0; t < 8; ++t) {
            const float bias = sB2[mrow + 16 * t];
#pragma unroll
            for (int r = 0; r < 8; ++r) {
                myStage[(r + 8 * kh) * SRS + (mrow + 16 * t)]      = acc2L[t][r] + bias;
                myStage[(16 + r + 8 * kh) * SRS + (mrow + 16 * t)] = acc2H[t][r] + bias;
            }
        }
        asm volatile("s_wait_dscnt 0x0" ::: "memory");

        // ---- LayerNorm + residual + store: one lane per row, row cached in regs ----
        const int mo = lane;                       // 0..31
        const float* rowS = myStage + mo * SRS;
        v4f buf[32];
        float sum = 0.f, ss = 0.f;
#pragma unroll
        for (int i = 0; i < 32; ++i) {
            buf[i] = *(const v4f*)(rowS + 4 * i);
#pragma unroll
            for (int j = 0; j < 4; ++j) { sum += buf[i][j]; ss += buf[i][j] * buf[i][j]; }
        }
        const float mu  = sum * (1.0f / 128.0f);
        const float var = ss * (1.0f / 128.0f) - mu * mu;
        const float rs  = rsqrtf(var + 1e-5f);

        const long long eo = eBase + mo;
        if (eo < nE) {
            const float* ef = efeat + eo * FD;
            float*       op = out   + eo * FD;
#pragma unroll
            for (int i = 0; i < 32; ++i) {
                v4f e4 = *(const v4f*)(ef + 4 * i);
                v4f g4 = *(const v4f*)(sG  + 4 * i);
                v4f b4 = *(const v4f*)(sBt + 4 * i);
                v4f o;
#pragma unroll
                for (int j = 0; j < 4; ++j) o[j] = (buf[i][j] - mu) * rs * g4[j] + b4[j] + e4[j];
                *(v4f*)(op + 4 * i) = o;
            }
        }
        asm volatile("s_wait_dscnt 0x0" ::: "memory");   // stage reads done before next tile's h writes
    }
}

__global__ void copy_nfeat_kernel(const float4* __restrict__ src,
                                  float4* __restrict__ dst, long long n4)
{
    long long i      = (long long)blockIdx.x * blockDim.x + threadIdx.x;
    long long stride = (long long)gridDim.x * blockDim.x;
    for (; i < n4; i += stride) dst[i] = src[i];
}

extern "C" void kernel_launch(void* const* d_in, const int* in_sizes, int n_in,
                              void* d_out, int out_size, void* d_ws, size_t ws_size,
                              hipStream_t stream) {
    const float* efeat   = (const float*)d_in[0];
    const float* nfeat   = (const float*)d_in[1];
    const int*   src_idx = (const int*)  d_in[2];
    const int*   dst_idx = (const int*)  d_in[3];
    const float* W1      = (const float*)d_in[4];
    const float* b1      = (const float*)d_in[5];
    const float* W2      = (const float*)d_in[6];
    const float* b2      = (const float*)d_in[7];
    const float* ln_g    = (const float*)d_in[8];
    const float* ln_b    = (const float*)d_in[9];

    const long long E = (long long)in_sizes[0] / FD;
    const long long N = (long long)in_sizes[1] / FD;
    float* out0 = (float*)d_out;            // [E,128] edge output
    float* out1 = out0 + E * FD;            // [N,128] passthrough nfeat

    const long long nTiles = (E + TILE_M - 1) / TILE_M;
    int grid = (int)(nTiles < 1024 ? nTiles : 1024);

    // 266KB dynamic LDS (> 64KB default): opt in every call (deterministic, capture-safe API call)
    (void)hipFuncSetAttribute(reinterpret_cast<const void*>(edge_mlp_wmma_kernel),
                              hipFuncAttributeMaxDynamicSharedMemorySize, SM_TOTAL);

    edge_mlp_wmma_kernel<<<grid, 256, SM_TOTAL, stream>>>(
        efeat, nfeat, src_idx, dst_idx, W1, b1, W2, b2, ln_g, ln_b, out0, E, nTiles);

    long long n4 = (N * FD) / 4;
    int cgrid = (int)((n4 + 255) / 256);
    if (cgrid > 2048) cgrid = 2048;
    if (cgrid < 1) cgrid = 1;
    copy_nfeat_kernel<<<cgrid, 256, 0, stream>>>((const float4*)nfeat, (float4*)out1, n4);
}
